// RetinaNet_TorchVision_36189394436908
// MI455X (gfx1250) — 8-bit matrix operand order repaired and checked by reading, compiled, not run on hardware
//
#include <hip/hip_runtime.h>
#include <hip/hip_bf16.h>
#include <math.h>

typedef unsigned int u32;
typedef unsigned char u8;
typedef __attribute__((ext_vector_type(8))) int v8i;

#define NB        2
#define NC        91
#define NLVL      5
#define TOPK      1000
#define MAXDET    300
#define KTOT      (NLVL * TOPK)
#define KP        5056
#define NT        (KP / 16)
#define A_TOTAL   200700
#define SCORE_THR 0.05f
#define IOU_THR   0.5f
#define W_IMG     1333.0f
#define H_IMG     800.0f
#define BBOX_CLIP 4.135166556742356f
#define SORTN     8192

__constant__ int d_lvl_off[5] = {0, 150300, 188100, 197550, 200007};
__constant__ int d_lvl_cnt[5] = {13677300, 3439800, 859950, 223587, 63063};

struct SelState { u32 need, prefix, gcount, cntG, cntE; };

__device__ inline u32 f2key(float f) {
  u32 u = __float_as_uint(f);
  return (u >> 31) ? ~u : (u | 0x80000000u);
}
__device__ inline float key2f(u32 k) {
  u32 u = (k & 0x80000000u) ? (k ^ 0x80000000u) : ~k;
  return __uint_as_float(u);
}

__global__ void k_init(u32* hist, SelState* st, float* candScore,
                       float* candBox, int* candCls) {
  int gid = blockIdx.x * blockDim.x + threadIdx.x;
  int stride = gridDim.x * blockDim.x;
  for (int i = gid; i < 10 * 65536; i += stride) hist[i] = 0;
  if (gid < 10) {
    st[gid].need = TOPK; st[gid].prefix = 0; st[gid].gcount = 0;
    st[gid].cntG = 0; st[gid].cntE = 0;
  }
  if (gid < NB * (KP - KTOT)) {
    int b = gid / (KP - KTOT);
    int s = KTOT + gid % (KP - KTOT);
    candScore[b * KP + s] = -1e30f;
    candCls[b * KP + s] = -1;
    for (int q = 0; q < 4; q++) candBox[(size_t)(b * KP + s) * 4 + q] = 0.f;
  }
}

__global__ void k_hist(const float* __restrict__ logits, u32* hist,
                       const SelState* st, int pass) {
  int pair = blockIdx.y;
  int b = pair / NLVL, lvl = pair % NLVL;
  const float* base =
      logits + (size_t)b * A_TOTAL * NC + (size_t)d_lvl_off[lvl] * NC;
  int N = d_lvl_cnt[lvl];
  u32 pfx = st[pair].prefix;
  u32* h = hist + (size_t)pair * 65536;
  for (int i = blockIdx.x * blockDim.x + threadIdx.x; i < N;
       i += gridDim.x * blockDim.x) {
    u32 key = f2key(base[i]);
    if (pass == 0) {
      atomicAdd(&h[key >> 16], 1u);
    } else if ((key >> 16) == pfx) {
      atomicAdd(&h[key & 0xFFFFu], 1u);
    }
  }
}

__global__ void k_select(u32* hist, SelState* st) {
  int pair = blockIdx.x;
  int tid = threadIdx.x;
  u32* h = hist + (size_t)pair * 65536;
  __shared__ u32 part[256];
  u32 s = 0;
  for (int k2 = 0; k2 < 256; k2++) s += h[tid * 256 + k2];
  part[tid] = s;
  __syncthreads();
  if (tid == 0) {
    u32 need = st[pair].need;
    u32 cum = 0;
    int Tb = 0;
    for (int x = 255; x >= 0; x--) {
      if (cum + part[x] >= need) { Tb = x; break; }
      cum += part[x];
    }
    int d = 0;
    for (int k2 = 255; k2 >= 0; k2--) {
      u32 c = h[Tb * 256 + k2];
      if (cum + c >= need) { d = k2; break; }
      cum += c;
    }
    u32 digit = (u32)(Tb * 256 + d);
    st[pair].gcount += cum;
    st[pair].need = need - cum;
    st[pair].prefix = (st[pair].prefix << 16) | digit;
  }
  __syncthreads();
  for (int k2 = 0; k2 < 256; k2++) h[tid * 256 + k2] = 0;
}

__global__ void k_collect(const float* __restrict__ logits, SelState* st,
                          u32* listKey, u32* listIdx) {
  int pair = blockIdx.y;
  int b = pair / NLVL, lvl = pair % NLVL;
  const float* base =
      logits + (size_t)b * A_TOTAL * NC + (size_t)d_lvl_off[lvl] * NC;
  int N = d_lvl_cnt[lvl];
  u32 T = st[pair].prefix;
  u32 G = st[pair].gcount;
  u32 need = st[pair].need;
  for (int i = blockIdx.x * blockDim.x + threadIdx.x; i < N;
       i += gridDim.x * blockDim.x) {
    u32 key = f2key(base[i]);
    if (key > T) {
      u32 slot = atomicAdd(&st[pair].cntG, 1u);
      if (slot < G) { listKey[pair * TOPK + slot] = key;
                      listIdx[pair * TOPK + slot] = (u32)i; }
    } else if (key == T) {
      u32 tq = atomicAdd(&st[pair].cntE, 1u);
      if (tq < need) { listKey[pair * TOPK + G + tq] = key;
                       listIdx[pair * TOPK + G + tq] = (u32)i; }
    }
  }
}

__global__ void k_decode(const u32* listKey, const u32* listIdx,
                         const float* __restrict__ reg,
                         const float* __restrict__ priors,
                         float* candScore, float* candBox, int* candCls) {
  int gid = blockIdx.x * blockDim.x + threadIdx.x;
  if (gid >= 10 * TOPK) return;
  int p = gid / TOPK, e = gid % TOPK;
  int b = p / NLVL, lvl = p % NLVL;
  u32 key = listKey[p * TOPK + e];
  u32 idx = listIdx[p * TOPK + e];
  float logit = key2f(key);
  float sc = 1.0f / (1.0f + expf(-logit));
  float sOut = (sc > SCORE_THR) ? sc : -1.0f;
  int aloc = (int)(idx / NC);
  int cls = (int)(idx - (u32)aloc * NC);
  int anchor = d_lvl_off[lvl] + aloc;
  const float* r = reg + ((size_t)b * A_TOTAL + anchor) * 4;
  const float* pr = priors + (size_t)anchor * 4;
  float pw = pr[2] - pr[0], ph = pr[3] - pr[1];
  float pcx = pr[0] + 0.5f * pw, pcy = pr[1] + 0.5f * ph;
  float cx = r[0] * pw + pcx, cy = r[1] * ph + pcy;
  float w = expf(fminf(r[2], BBOX_CLIP)) * pw;
  float h = expf(fminf(r[3], BBOX_CLIP)) * ph;
  float x1 = fminf(fmaxf(cx - 0.5f * w, 0.f), W_IMG);
  float y1 = fminf(fmaxf(cy - 0.5f * h, 0.f), H_IMG);
  float x2 = fminf(fmaxf(cx + 0.5f * w, 0.f), W_IMG);
  float y2 = fminf(fmaxf(cy + 0.5f * h, 0.f), H_IMG);
  int slot = lvl * TOPK + e;
  size_t o = (size_t)b * KP + slot;
  candScore[o] = sOut;
  candCls[o] = cls;
  candBox[o * 4 + 0] = x1; candBox[o * 4 + 1] = y1;
  candBox[o * 4 + 2] = x2; candBox[o * 4 + 3] = y2;
}

__global__ void k_sort(const float* candScore, const float* candBox,
                       const int* candCls, float* sScore, float* sBox,
                       int* sCls) {
  extern __shared__ char smem[];
  float* key = (float*)smem;
  int* val = (int*)(smem + SORTN * 4);
  int b = blockIdx.x;
  int tid = threadIdx.x, nt = blockDim.x;
  for (int s = tid; s < SORTN; s += nt) {
    if (s < KP) { key[s] = candScore[(size_t)b * KP + s]; val[s] = s; }
    else        { key[s] = -1e38f; val[s] = -1; }
  }
  __syncthreads();
  for (int ksz = 2; ksz <= SORTN; ksz <<= 1) {
    for (int j = ksz >> 1; j > 0; j >>= 1) {
      for (int i = tid; i < SORTN; i += nt) {
        int l = i ^ j;
        if (l > i) {
          bool desc = ((i & ksz) == 0);
          float a = key[i], c = key[l];
          bool sw = desc ? (a < c) : (a > c);
          if (sw) {
            key[i] = c; key[l] = a;
            int t = val[i]; val[i] = val[l]; val[l] = t;
          }
        }
      }
      __syncthreads();
    }
  }
  for (int s = tid; s < KP; s += nt) {
    int o = val[s];
    size_t d = (size_t)b * KP + s;
    if (o >= 0) {
      size_t so = (size_t)b * KP + o;
      sScore[d] = key[s];
      sCls[d] = candCls[so];
      for (int q = 0; q < 4; q++) sBox[d * 4 + q] = candBox[so * 4 + q];
    } else {
      sScore[d] = -1e38f; sCls[d] = -1;
      for (int q = 0; q < 4; q++) sBox[d * 4 + q] = 0.f;
    }
  }
}

__global__ void k_adj(const float* __restrict__ sBox,
                      const int* __restrict__ sCls, u8* adj) {
  int b = blockIdx.z;
  int i = blockIdx.y * 16 + threadIdx.y;
  int j = blockIdx.x * 16 + threadIdx.x;
  if (i >= KP || j >= KP) return;
  u8 out = 0;
  int ci = sCls[(size_t)b * KP + i], cj = sCls[(size_t)b * KP + j];
  if (i != j && ci >= 0 && ci == cj) {
    const float* bi = sBox + ((size_t)b * KP + i) * 4;
    const float* bj = sBox + ((size_t)b * KP + j) * 4;
    float ai = (bi[2] - bi[0]) * (bi[3] - bi[1]);
    float aj = (bj[2] - bj[0]) * (bj[3] - bj[1]);
    float iw = fmaxf(fminf(bi[2], bj[2]) - fmaxf(bi[0], bj[0]), 0.f);
    float ih = fmaxf(fminf(bi[3], bj[3]) - fmaxf(bi[1], bj[1]), 0.f);
    float inter = iw * ih;
    float iou = inter / (ai + aj - inter + 1e-9f);
    out = (iou > IOU_THR) ? 1 : 0;
  }
  adj[(size_t)b * KP * KP + (size_t)i * KP + j] = out;
}

__global__ void k_nms(const u8* __restrict__ adj,
                      const float* __restrict__ sScore, u8* keepOut) {
  int b = blockIdx.x;
  int tid = threadIdx.x;
  int lane = tid & 31;
  int wave = tid >> 5;
  int nwave = blockDim.x >> 5;
  __shared__ int supp[KP];
  __shared__ u8 keepSh[KP];
  __shared__ u8 validSh[KP];
  const u8* A = adj + (size_t)b * KP * KP;

  for (int i = tid; i < KP; i += blockDim.x) {
    supp[i] = 0;
    keepSh[i] = 0;
    validSh[i] = (i < KTOT) && (sScore[(size_t)b * KP + i] > SCORE_THR);
  }
  __syncthreads();

  for (int t = 0; t < NT; t++) {
    int cb = t * 16;
    if (tid == 0) {
      for (int r2 = 0; r2 < 16; r2++) {
        int i = cb + r2;
        int s = supp[i];
        const u8* arow = A + (size_t)i * KP + cb;
        for (int q = 0; q < r2; q++) s += (int)(keepSh[cb + q] & arow[q]);
        keepSh[i] = (validSh[i] && s == 0) ? 1 : 0;
      }
    }
    __syncthreads();

    for (int chunk = t + 1 + wave; chunk < NT; chunk += nwave) {
      int m0 = chunk * 16;
      int M = lane & 15;
      int kb = (lane < 16) ? 0 : 8;
      const u8* arow = A + (size_t)(m0 + M) * KP + cb;
      v8i av = {0, 0, 0, 0, 0, 0, 0, 0};
      av[0] = *(const int*)(arow + kb);
      av[1] = *(const int*)(arow + kb + 4);
      v8i bv = {0, 0, 0, 0, 0, 0, 0, 0};
      if ((lane & 15) == 0) {
#pragma unroll
        for (int v = 0; v < 2; v++) {
          u32 w = (u32)keepSh[cb + kb + 4 * v] |
                  ((u32)keepSh[cb + kb + 4 * v + 1] << 8) |
                  ((u32)keepSh[cb + kb + 4 * v + 2] << 16) |
                  ((u32)keepSh[cb + kb + 4 * v + 3] << 24);
          bv[v] = (int)w;
        }
      }
      v8i cv = {0, 0, 0, 0, 0, 0, 0, 0};
      cv = __builtin_amdgcn_wmma_i32_16x16x64_iu8(false, av, false, bv, cv,
                                                  false, false);
      if ((lane & 15) == 0) {
        int mb = m0 + (lane >> 4) * 8;
#pragma unroll
        for (int r2 = 0; r2 < 8; r2++) supp[mb + r2] += cv[r2];
      }
    }
    __syncthreads();
  }

  for (int i = tid; i < KP; i += blockDim.x)
    keepOut[(size_t)b * KP + i] = keepSh[i];
}

__global__ void k_final(const u8* keep, const float* sScore,
                        const float* sBox, const int* sCls, float* out) {
  int b = blockIdx.x;
  if (threadIdx.x != 0) return;
  float* ob = out + (size_t)b * MAXDET * 4;
  float* os = out + (size_t)NB * MAXDET * 4 + (size_t)b * MAXDET;
  float* oc = out + (size_t)NB * MAXDET * 5 + (size_t)b * MAXDET;
  int cnt = 0;
  for (int i = 0; i < KTOT && cnt < MAXDET; i++) {
    if (keep[(size_t)b * KP + i]) {
      size_t so = (size_t)b * KP + i;
      for (int q = 0; q < 4; q++) ob[cnt * 4 + q] = sBox[so * 4 + q];
      os[cnt] = sScore[so];
      oc[cnt] = (float)sCls[so];
      cnt++;
    }
  }
  for (; cnt < MAXDET; cnt++) {
    for (int q = 0; q < 4; q++) ob[cnt * 4 + q] = 0.f;
    os[cnt] = 0.f;
    oc[cnt] = -1.f;
  }
}

extern "C" void kernel_launch(void* const* d_in, const int* in_sizes, int n_in,
                              void* d_out, int out_size, void* d_ws,
                              size_t ws_size, hipStream_t stream) {
  const float* reg    = (const float*)d_in[0];
  const float* logits = (const float*)d_in[1];
  const float* priors = (const float*)d_in[2];
  (void)in_sizes; (void)n_in; (void)out_size; (void)ws_size;

  char* w = (char*)d_ws;
  size_t off = 0;
  auto alloc = [&](size_t bytes) {
    char* p = w + off;
    off = (off + bytes + 255) & ~(size_t)255;
    return p;
  };
  u32*      hist      = (u32*)alloc((size_t)10 * 65536 * 4);
  SelState* st        = (SelState*)alloc(10 * sizeof(SelState));
  u32*      listKey   = (u32*)alloc((size_t)10 * TOPK * 4);
  u32*      listIdx   = (u32*)alloc((size_t)10 * TOPK * 4);
  float*    candScore = (float*)alloc((size_t)NB * KP * 4);
  float*    candBox   = (float*)alloc((size_t)NB * KP * 4 * 4);
  int*      candCls   = (int*)alloc((size_t)NB * KP * 4);
  float*    sScore    = (float*)alloc((size_t)NB * KP * 4);
  float*    sBox      = (float*)alloc((size_t)NB * KP * 4 * 4);
  int*      sCls      = (int*)alloc((size_t)NB * KP * 4);
  u8*       keepOut   = (u8*)alloc((size_t)NB * KP);
  u8*       adj       = (u8*)alloc((size_t)NB * KP * KP);

  k_init<<<1024, 256, 0, stream>>>(hist, st, candScore, candBox, candCls);
  k_hist<<<dim3(512, 10), 256, 0, stream>>>(logits, hist, st, 0);
  k_select<<<10, 256, 0, stream>>>(hist, st);
  k_hist<<<dim3(512, 10), 256, 0, stream>>>(logits, hist, st, 1);
  k_select<<<10, 256, 0, stream>>>(hist, st);
  k_collect<<<dim3(512, 10), 256, 0, stream>>>(logits, st, listKey, listIdx);
  k_decode<<<(10 * TOPK + 255) / 256, 256, 0, stream>>>(
      listKey, listIdx, reg, priors, candScore, candBox, candCls);
  k_sort<<<NB, 1024, SORTN * 8, stream>>>(candScore, candBox, candCls, sScore,
                                          sBox, sCls);
  k_adj<<<dim3(NT, NT, NB), dim3(16, 16), 0, stream>>>(sBox, sCls, adj);
  k_nms<<<NB, 256, 0, stream>>>(adj, sScore, keepOut);
  k_final<<<NB, 32, 0, stream>>>(keepOut, sScore, sBox, sCls, (float*)d_out);
}
